// Head_76020921139706
// MI455X (gfx1250) — matrix operand order repaired and checked by reading, not compiled, not run on hardware
//
#include <hip/hip_runtime.h>
#include <hip/hip_bf16.h>

typedef _Float16 v8h  __attribute__((ext_vector_type(8)));
typedef _Float16 v16h __attribute__((ext_vector_type(16)));
typedef float    v8f  __attribute__((ext_vector_type(8)));

#define BB 8
#define TT 2048
#define CC 1024
#define HH 64
#define LDA 1032   // padded LDS row stride (halves) for x tile
#define PST 40     // padded LDS row stride (halves) for P tile
#define KST 72     // padded LDS row stride (halves) for staged K tile (144B rows)
#define VST 40     // padded LDS row stride (halves) for staged V tile (80B rows)

__device__ __forceinline__ v8f wmma16(v16h a, v16h b, v8f c) {
  return __builtin_amdgcn_wmma_f32_16x16x32_f16(false, a, false, b, (short)0, c,
                                                false, false);
}

__device__ __forceinline__ v16h cat8(v8h lo, v8h hi) {
  return __builtin_shufflevector(lo, hi, 0, 1, 2, 3, 4, 5, 6, 7,
                                 8, 9, 10, 11, 12, 13, 14, 15);
}

__device__ __forceinline__ unsigned lds_off(const void* p) {
  // LDS flat pointers carry the byte offset in the low 32 bits (aperture hi bits)
  return (unsigned)(unsigned long long)p;
}

// ---------------------------------------------------------------------------
// Kernel 0: W [C,H] f32  ->  Wt [3][H][C] f16 (transposed, WMMA-B friendly)
// ---------------------------------------------------------------------------
__global__ __launch_bounds__(256) void wtrans_kernel(
    const float* __restrict__ Wq, const float* __restrict__ Wk,
    const float* __restrict__ Wv, _Float16* __restrict__ Wt) {
  int tid = blockIdx.x * 256 + threadIdx.x;   // < 3*64*1024
  int m = tid >> 16;
  int r = tid & 65535;
  int n = r >> 10;     // head column 0..63
  int k = r & 1023;    // C index
  const float* W = (m == 0) ? Wq : ((m == 1) ? Wk : Wv);
  Wt[tid] = (_Float16)W[k * HH + n];
}

// ---------------------------------------------------------------------------
// Kernel 1: fused QKV projection. One 16-row x tile per block, 4 waves each
// own 16 output columns of q/k/v. A-frag from LDS, B-frags from L2-resident Wt.
// Q,K stored [B*T][64] f16; V stored transposed [B][64][T] f16.
// ---------------------------------------------------------------------------
__global__ __launch_bounds__(128) void qkv_kernel(
    const float* __restrict__ x, const _Float16* __restrict__ Wt,
    _Float16* __restrict__ Qh, _Float16* __restrict__ Kh,
    _Float16* __restrict__ Vt) {
  __shared__ _Float16 xs[16 * LDA];
  const int tid = threadIdx.x;
  const long m0 = (long)blockIdx.x * 16;

  // ---- stage x tile (16 x 1024 f32) into LDS as f16 ----
  const float* xb = x + m0 * CC;
#pragma unroll 4
  for (int i = 0; i < 32; ++i) {
    int f   = (i * 128 + tid) * 4;
    int row = f >> 10;
    int col = f & 1023;
    float4 v = *(const float4*)(xb + row * CC + col);
    _Float16* d = &xs[row * LDA + col];
    d[0] = (_Float16)v.x; d[1] = (_Float16)v.y;
    d[2] = (_Float16)v.z; d[3] = (_Float16)v.w;
  }
  __syncthreads();

  const int w = tid >> 5, lane = tid & 31;
  const int n = lane & 15, g = lane >> 4;
  const int col = w * 16 + n;

  const _Float16* wqp = Wt + 0 * (HH * CC) + col * CC;
  const _Float16* wkp = Wt + 1 * (HH * CC) + col * CC;
  const _Float16* wvp = Wt + 2 * (HH * CC) + col * CC;

  v8f acc0 = {}, acc1 = {}, acc2 = {};

#pragma unroll 4
  for (int kk = 0; kk < 32; ++kk) {
    // A fragment (16x32 f16): row=n holds K in {16g..16g+15} (contiguous)
    const _Float16* ap = &xs[n * LDA + kk * 32];
    v16h a = cat8(*(const v8h*)(ap + 16 * g), *(const v8h*)(ap + 16 * g + 8));
    // B fragments: lane column = col, K = kk*32 + 16g + e (contiguous halves)
    v16h bq = *(const v16h*)(wqp + kk * 32 + 16 * g);
    v16h bk = *(const v16h*)(wkp + kk * 32 + 16 * g);
    v16h bv = *(const v16h*)(wvp + kk * 32 + 16 * g);
    acc0 = wmma16(a, bq, acc0);
    acc1 = wmma16(a, bk, acc1);
    acc2 = wmma16(a, bv, acc2);
  }

  // ---- store Q, K row-major [B*T][64] ----
#pragma unroll
  for (int e = 0; e < 8; ++e) {
    long trow = m0 + 8 * g + e;        // C/D layout: row M = e + 8g
    Qh[trow * HH + col] = (_Float16)acc0[e];
    Kh[trow * HH + col] = (_Float16)acc1[e];
  }
  // ---- store V transposed [B][64][T]: 8 consecutive tokens -> one 16B store
  long bidx = m0 / TT;
  long tin  = (m0 % TT) + 8 * g;
  v8h vv;
#pragma unroll
  for (int e = 0; e < 8; ++e) vv[e] = (_Float16)acc2[e];
  *(v8h*)(Vt + (bidx * HH + col) * TT + tin) = vv;
}

// ---------------------------------------------------------------------------
// Async staging helpers: one instruction copies 32 lanes x 16B = 512B.
// Tracked with ASYNCcnt; per-wave double buffers.
// ---------------------------------------------------------------------------
__device__ __forceinline__ void stage_k_tile(const _Float16* Kb, int kb,
                                             const _Float16* dst, int lane) {
  // global: 32 rows x 128B (row = key); LDS: stride 144B. 8 instrs, 4 rows each.
  unsigned row   = (unsigned)lane >> 3;   // 0..3
  unsigned chunk = (unsigned)lane & 7;    // 0..7 (16B chunks)
  unsigned l0 = lds_off(dst) + row * (KST * 2) + chunk * 16u;
  unsigned g0 = (unsigned)(kb + (int)row) * (HH * 2) + chunk * 16u;
#pragma unroll
  for (int i = 0; i < 8; ++i) {
    asm volatile("global_load_async_to_lds_b128 %0, %1, %2"
                 :: "v"(l0 + (unsigned)(i * 4 * KST * 2)),
                    "v"(g0 + (unsigned)(i * 4 * HH * 2)),
                    "s"(Kb)
                 : "memory");
  }
}

__device__ __forceinline__ void stage_v_tile(const _Float16* Vb, int kb,
                                             const _Float16* dst, int lane) {
  // global: 64 rows (head dims) x 64B at stride T*2; LDS: stride 80B.
  unsigned row   = (unsigned)lane >> 2;   // 0..7
  unsigned chunk = (unsigned)lane & 3;    // 0..3 (16B chunks)
  unsigned l0 = lds_off(dst) + row * (VST * 2) + chunk * 16u;
  unsigned g0 = row * (TT * 2) + chunk * 16u + (unsigned)(kb * 2);
#pragma unroll
  for (int i = 0; i < 8; ++i) {
    asm volatile("global_load_async_to_lds_b128 %0, %1, %2"
                 :: "v"(l0 + (unsigned)(i * 8 * VST * 2)),
                    "v"(g0 + (unsigned)(i * 8 * TT * 2)),
                    "s"(Vb)
                 : "memory");
  }
}

// ---------------------------------------------------------------------------
// Kernel 2: causal flash attention. One wave per 16-query tile, 32-key tiles.
// K/V tiles double-buffered in LDS via async-to-LDS; online softmax; P bounced
// through LDS to convert C->A fragment layout.
// ---------------------------------------------------------------------------
__global__ __launch_bounds__(128) void attn_kernel(
    const _Float16* __restrict__ Qh, const _Float16* __restrict__ Kh,
    const _Float16* __restrict__ Vt, float* __restrict__ out) {
  __shared__ _Float16 kbuf[4][2][32 * KST];
  __shared__ _Float16 vbuf[4][2][64 * VST];
  __shared__ _Float16 pls[4][16 * PST];
  const int tid = threadIdx.x;
  const int w = tid >> 5, lane = tid & 31;
  const int n = lane & 15, g = lane >> 4;
  const int b  = blockIdx.y;
  const int q0 = blockIdx.x * 64 + w * 16;

  const _Float16* Qb = Qh + (long)b * TT * HH;
  const _Float16* Kb = Kh + (long)b * TT * HH;
  const _Float16* Vb = Vt + (long)b * HH * TT;
  _Float16* P = &pls[w][0];

  // Q A-fragments for head-dim k-steps [0,32) and [32,64)
  const _Float16* qr = Qb + (q0 + n) * HH;
  v16h aq0 = *(const v16h*)(qr + 16 * g);
  v16h aq1 = *(const v16h*)(qr + 32 + 16 * g);

  float mrow[8], lrow[8];
#pragma unroll
  for (int e = 0; e < 8; ++e) { mrow[e] = -3.0e38f; lrow[e] = 0.0f; }
  v8f acc[4] = {{}, {}, {}, {}};

  const int ntiles = (q0 + 16 + 31) / 32;   // causal: keys < q0+16

  // prologue: stage tile 0 into buffer 0
  stage_k_tile(Kb, 0, kbuf[w][0], lane);
  stage_v_tile(Vb, 0, vbuf[w][0], lane);

  for (int j = 0; j < ntiles; ++j) {
    const int kb = j * 32;
    if (j + 2 < ntiles)   // global_prefetch_b8 two tiles ahead into L2
      __builtin_prefetch(Kb + (kb + 64 + n) * HH, 0, 0);

    // stage next tile into the other buffer, then wait for current tile
    if (j + 1 < ntiles) {
      stage_k_tile(Kb, kb + 32, kbuf[w][(j + 1) & 1], lane);
      stage_v_tile(Vb, kb + 32, vbuf[w][(j + 1) & 1], lane);
      asm volatile("s_wait_asynccnt 0x10" ::: "memory");  // <=16: tile j done
    } else {
      asm volatile("s_wait_asynccnt 0x0" ::: "memory");
    }

    const _Float16* kt = kbuf[w][j & 1];
    const _Float16* vt = vbuf[w][j & 1];

    // ---- S = Q K^T for 32 keys (two 16x16 C tiles) ----
    const _Float16* k0p = kt + n * KST + 16 * g;
    const _Float16* k1p = kt + (16 + n) * KST + 16 * g;
    v16h bk00 = *(const v16h*)(k0p);
    v16h bk01 = *(const v16h*)(k0p + 32);
    v16h bk10 = *(const v16h*)(k1p);
    v16h bk11 = *(const v16h*)(k1p + 32);
    v8f s0 = {}, s1 = {};
    s0 = wmma16(aq0, bk00, s0); s0 = wmma16(aq1, bk01, s0);
    s1 = wmma16(aq0, bk10, s1); s1 = wmma16(aq1, bk11, s1);

    // ---- online softmax (row M = e + 8g, col N = lane&15) ----
#pragma unroll
    for (int e = 0; e < 8; ++e) {
      const int mq = q0 + 8 * g + e;
      float v0 = (kb + n      <= mq) ? s0[e] * 0.125f : -3.0e38f;
      float v1 = (kb + 16 + n <= mq) ? s1[e] * 0.125f : -3.0e38f;
      float rm = fmaxf(v0, v1);
#pragma unroll
      for (int msk = 1; msk < 16; msk <<= 1) rm = fmaxf(rm, __shfl_xor(rm, msk, 32));
      float mnew  = fmaxf(mrow[e], rm);
      float p0    = __expf(v0 - mnew);
      float p1    = __expf(v1 - mnew);
      float rs    = p0 + p1;
#pragma unroll
      for (int msk = 1; msk < 16; msk <<= 1) rs += __shfl_xor(rs, msk, 32);
      float alpha = __expf(mrow[e] - mnew);
      mrow[e] = mnew;
      lrow[e] = lrow[e] * alpha + rs;
      acc[0][e] *= alpha; acc[1][e] *= alpha;
      acc[2][e] *= alpha; acc[3][e] *= alpha;
      const int prow = 8 * g + e;
      P[prow * PST + n]      = (_Float16)p0;
      P[prow * PST + 16 + n] = (_Float16)p1;
    }
    // same-wave LDS ops are in-order (DScnt); fence the compiler + hw counter
    asm volatile("" ::: "memory");
    asm volatile("s_wait_dscnt 0" ::: "memory");

    // ---- P (16x32) as A-fragment, O += P * V ----
    const _Float16* pr = P + n * PST;
    v16h ap = cat8(*(const v8h*)(pr + 16 * g), *(const v8h*)(pr + 16 * g + 8));
#pragma unroll
    for (int t = 0; t < 4; ++t) {
      v16h bv = *(const v16h*)(vt + (t * 16 + n) * VST + 16 * g);
      acc[t] = wmma16(ap, bv, acc[t]);
    }
  }

  // ---- normalize and write out [B][T][64] f32 ----
  float* ob = out + ((long)b * TT + q0) * HH;
#pragma unroll
  for (int e = 0; e < 8; ++e) {
    float inv = 1.0f / lrow[e];
    int r = 8 * g + e;
#pragma unroll
    for (int t = 0; t < 4; ++t)
      ob[r * HH + t * 16 + n] = acc[t][e] * inv;
  }
}

// ---------------------------------------------------------------------------
extern "C" void kernel_launch(void* const* d_in, const int* in_sizes, int n_in,
                              void* d_out, int out_size, void* d_ws, size_t ws_size,
                              hipStream_t stream) {
  const float* x  = (const float*)d_in[0];
  const float* Wq = (const float*)d_in[1];
  const float* Wk = (const float*)d_in[2];
  const float* Wv = (const float*)d_in[3];
  float* out = (float*)d_out;

  char* ws = (char*)d_ws;
  const size_t WT_BYTES  = (size_t)3 * HH * CC * 2;       // 384 KB
  const size_t QKV_BYTES = (size_t)BB * TT * HH * 2;      // 2 MB each
  _Float16* Wt = (_Float16*)(ws);
  _Float16* Qh = (_Float16*)(ws + WT_BYTES);
  _Float16* Kh = (_Float16*)(ws + WT_BYTES + QKV_BYTES);
  _Float16* Vt = (_Float16*)(ws + WT_BYTES + 2 * QKV_BYTES);

  wtrans_kernel<<<(3 * HH * CC) / 256, 256, 0, stream>>>(Wq, Wk, Wv, Wt);
  qkv_kernel<<<(BB * TT) / 16, 128, 0, stream>>>(x, Wt, Qh, Kh, Vt);
  dim3 g2(TT / 64, BB);
  attn_kernel<<<g2, 128, 0, stream>>>(Qh, Kh, Vt, out);
}
